// MPNEncoder_37142877176027
// MI455X (gfx1250) — compile-verified, matrix operand order checked by reading, not run on hardware
//
#include <hip/hip_runtime.h>

// ---------------------------------------------------------------------------
// MPNN (chemprop D-MPNN) encoder for MI455X / gfx1250.
// GEMMs run on v_wmma_f32_16x16x32_bf16 (f32 accumulate); bond messages are
// stored bf16 (stride 320, K padded with zeros) so the gather passes stay
// L2-resident (128 MB < 192 MB L2). Residual b_input kept f32.
// Inner GEMM loop batches all B-fragment LDS loads per K-step so WMMAs issue
// back-to-back without per-op s_wait_dscnt stalls.
// ---------------------------------------------------------------------------

typedef __attribute__((ext_vector_type(16))) __bf16 v16bf;
typedef __attribute__((ext_vector_type(8)))  float  v8f;

union AFrag {
  uint4 u4[2];
  v16bf bf;
};

__device__ __forceinline__ unsigned short f2bf(float f) {
  unsigned int u = __float_as_uint(f);
  u += 0x7FFFu + ((u >> 16) & 1u);          // round-to-nearest-even
  return (unsigned short)(u >> 16);
}
__device__ __forceinline__ float bf2f(unsigned int h) {
  return __uint_as_float(h << 16);
}

#define N_ATOMS 100000
#define N_BONDS 200000
#define NAT (N_ATOMS + 1)
#define NBD (N_BONDS + 1)
#define HID 300
#define MP 200064   // 1563 * 128, padded bond rows
#define AP 100096   // 782 * 128, padded atom rows
#define NUM_MOLS 2048

// --- pack a (K x 300) f32 weight matrix into WMMA-B fragment-major bf16 ----
// frag layout per (ks, nt): 32 lanes x 16 halfs contiguous; (L,h) -> k,n per
// ISA 7.12.2 16-bit B layout. K padded to KT*32, N padded to 320, zeros.
__global__ void pack_w_kernel(const float* __restrict__ W,
                              unsigned short* __restrict__ frags,
                              int K, int KT) {
  int idx = blockIdx.x * 256 + threadIdx.x;
  int total = KT * 20 * 512;
  if (idx >= total) return;
  int f = idx >> 9;
  int w = idx & 511;
  int L = w >> 4, h = w & 15;
  int ks = f / 20, nt = f % 20;
  int kl = h + (h >= 8 ? 8 : 0) + (L >= 16 ? 8 : 0);
  int k = ks * 32 + kl;
  int n = nt * 16 + (L & 15);
  float v = (k < K && n < HID) ? W[(size_t)k * HID + n] : 0.f;
  frags[idx] = f2bf(v);
}

// --- f_bonds (NBD x 147 f32) -> padded bf16 (stride 160) -------------------
__global__ void cvt_bonds_kernel(const float* __restrict__ fb,
                                 unsigned short* __restrict__ out) {
  int idx = blockIdx.x * 256 + threadIdx.x;
  if (idx >= NBD * 160) return;
  int row = idx / 160, k = idx % 160;
  float v = (k < 147) ? fb[(size_t)row * 147 + k] : 0.f;
  out[(size_t)row * 160 + k] = f2bf(v);
}

// --- a_message[a] = sum_j bmsg[a2b[a][j]]  (f32 out, stride 300) -----------
__global__ void agg_kernel(const unsigned short* __restrict__ bmsg,
                           const int* __restrict__ a2b,
                           float* __restrict__ amsg) {
  int idx = blockIdx.x * 256 + threadIdx.x;
  if (idx >= NAT * 75) return;
  int atom = idx / 75;
  int c = (idx % 75) * 4;
  float4 s = make_float4(0.f, 0.f, 0.f, 0.f);
#pragma unroll
  for (int j = 0; j < 6; j++) {
    int b = a2b[atom * 6 + j];
    uint2 v = *(const uint2*)(bmsg + (size_t)b * 320 + c);
    s.x += bf2f(v.x & 0xffffu);
    s.y += bf2f(v.x >> 16);
    s.z += bf2f(v.y & 0xffffu);
    s.w += bf2f(v.y >> 16);
  }
  *(float4*)(amsg + (size_t)atom * 300 + c) = s;
}

// --- hpre[b] = bf16( amsg[b2a[b]] - bmsg[b2revb[b]] ), zero-padded cols ----
__global__ void bond_pre_kernel(const float* __restrict__ amsg,
                                const unsigned short* __restrict__ bmsg,
                                const int* __restrict__ b2a,
                                const int* __restrict__ b2revb,
                                unsigned short* __restrict__ hpre) {
  int idx = blockIdx.x * 256 + threadIdx.x;
  if (idx >= NBD * 80) return;
  int bond = idx / 80;
  int c = (idx % 80) * 4;
  uint2 outv;
  if (c < 300) {
    int a = b2a[bond], rb = b2revb[bond];
    float4 am = *(const float4*)(amsg + (size_t)a * 300 + c);
    uint2 bm = *(const uint2*)(bmsg + (size_t)rb * 320 + c);
    float x0 = am.x - bf2f(bm.x & 0xffffu);
    float x1 = am.y - bf2f(bm.x >> 16);
    float x2 = am.z - bf2f(bm.y & 0xffffu);
    float x3 = am.w - bf2f(bm.y >> 16);
    outv.x = (unsigned int)f2bf(x0) | ((unsigned int)f2bf(x1) << 16);
    outv.y = (unsigned int)f2bf(x2) | ((unsigned int)f2bf(x3) << 16);
  } else {
    outv.x = 0u; outv.y = 0u;
  }
  *(uint2*)(hpre + (size_t)bond * 320 + c) = outv;
}

// --- a_in = bf16 concat(f_atoms, a_message), stride 448, zero pad ----------
__global__ void build_ain_kernel(const float* __restrict__ fa,
                                 const float* __restrict__ amsg,
                                 unsigned short* __restrict__ ain) {
  int idx = blockIdx.x * 256 + threadIdx.x;
  if (idx >= NAT * 448) return;
  int row = idx / 448, k = idx % 448;
  float v;
  if (k < 133)      v = fa[(size_t)row * 133 + k];
  else if (k < 433) v = amsg[(size_t)row * 300 + (k - 133)];
  else              v = 0.f;
  ain[(size_t)row * 448 + k] = f2bf(v);
}

// --- WMMA GEMM: C[M x 300] = A[M x KT*32] @ W, fused epilogues -------------
// MODE 0: outF = A@W (f32),       outB = bf16(relu(A@W))
// MODE 1: outB = bf16(relu(Cin + A@W))
// MODE 2: atomicAdd into outF[mol_ids[m-1]] of relu(A@W + Cin[bias]) (m>=1)
template <int MODE, int KT, int NTB>
__global__ __launch_bounds__(256) void gemm_kernel(
    const unsigned short* __restrict__ A, int strideA,
    const unsigned short* __restrict__ Wfr,
    const float* __restrict__ Cin,
    float* __restrict__ outF,
    unsigned short* __restrict__ outB,
    const int* __restrict__ molIds,
    int Mreal) {
  __shared__ unsigned short ldsW[KT * NTB * 512];
  const int tid = threadIdx.x;
  const int ntBase = blockIdx.y * NTB;

  // stage this block's N-group weight fragments into LDS
  for (int i = tid; i < KT * NTB * 64; i += 256) {
    int frag = i >> 6, q = i & 63;
    int ks = frag / NTB, ntl = frag % NTB;
    ((uint4*)ldsW)[(size_t)frag * 64 + q] =
        ((const uint4*)Wfr)[(size_t)(ks * 20 + ntBase + ntl) * 64 + q];
  }
  __syncthreads();

  const int wave = tid >> 5, lane = tid & 31;
  const int Mbase = (blockIdx.x * 8 + wave) * 16;
  const int laneHi = lane >> 4;   // 0: K-halves {0..7,16..23}; 1: {8..15,24..31}
  const int ln = lane & 15;

  v8f acc[NTB];
  const v8f vzero = {0.f, 0.f, 0.f, 0.f, 0.f, 0.f, 0.f, 0.f};
#pragma unroll
  for (int t = 0; t < NTB; t++) acc[t] = vzero;

  const unsigned short* arow =
      A + (size_t)(Mbase + ln) * strideA + laneHi * 8;

#pragma unroll
  for (int ks = 0; ks < KT; ks++) {
    // A fragment for this 16-row strip, K-step ks (two global b128 loads)
    AFrag a;
    a.u4[0] = *(const uint4*)(arow + ks * 32);
    a.u4[1] = *(const uint4*)(arow + ks * 32 + 16);
    // batch-load ALL B fragments for this K-step first (one ds clause),
    // then issue the NTB WMMAs back-to-back with a single dscnt wait.
    AFrag b[NTB];
#pragma unroll
    for (int nt = 0; nt < NTB; nt++) {
      const unsigned short* bp = ldsW + (((ks * NTB + nt) * 32 + lane) << 4);
      b[nt].u4[0] = *(const uint4*)(bp);
      b[nt].u4[1] = *(const uint4*)(bp + 8);
    }
#pragma unroll
    for (int nt = 0; nt < NTB; nt++) {
      acc[nt] = __builtin_amdgcn_wmma_f32_16x16x32_bf16(
          false, a.bf, false, b[nt].bf, (short)0, acc[nt], false, false);
    }
  }

#pragma unroll
  for (int nt = 0; nt < NTB; nt++) {
    int n = (ntBase + nt) * 16 + ln;
#pragma unroll
    for (int r = 0; r < 8; r++) {
      int m = Mbase + laneHi * 8 + r;
      float v = acc[nt][r];
      if constexpr (MODE == 0) {
        if (m < Mreal && n < HID) {
          outF[(size_t)m * HID + n] = v;
          outB[(size_t)m * 320 + n] = f2bf(fmaxf(v, 0.f));
        }
      } else if constexpr (MODE == 1) {
        if (m < Mreal && n < HID) {
          float pre = v + Cin[(size_t)m * HID + n];
          outB[(size_t)m * 320 + n] = f2bf(fmaxf(pre, 0.f));
        }
      } else {
        if (m >= 1 && m < Mreal && n < HID) {
          float h = fmaxf(v + Cin[n], 0.f);
          atomicAdd(outF + (size_t)molIds[m - 1] * HID + n, h);
        }
      }
    }
  }
}

__global__ void zero_kernel(float* p, int n) {
  int i = blockIdx.x * 256 + threadIdx.x;
  if (i < n) p[i] = 0.f;
}
__global__ void count_kernel(const int* __restrict__ mol, int* __restrict__ cnt) {
  int i = blockIdx.x * 256 + threadIdx.x;
  if (i < N_ATOMS) atomicAdd(cnt + mol[i], 1);
}
__global__ void finalize_kernel(float* out, const int* __restrict__ cnt) {
  int i = blockIdx.x * 256 + threadIdx.x;
  if (i < NUM_MOLS * HID) {
    int m = i / HID;
    out[i] = out[i] / fmaxf((float)cnt[m], 1.f);
  }
}

extern "C" void kernel_launch(void* const* d_in, const int* in_sizes, int n_in,
                              void* d_out, int out_size, void* d_ws, size_t ws_size,
                              hipStream_t stream) {
  const float* f_atoms = (const float*)d_in[0];
  const float* f_bonds = (const float*)d_in[1];
  const float* W_i     = (const float*)d_in[2];
  const float* W_h     = (const float*)d_in[3];
  const float* W_o     = (const float*)d_in[4];
  const float* b_o     = (const float*)d_in[5];
  const int* a2b       = (const int*)d_in[6];
  const int* b2a       = (const int*)d_in[7];
  const int* b2revb    = (const int*)d_in[8];
  const int* mol_ids   = (const int*)d_in[9];
  float* out = (float*)d_out;
  char* ws = (char*)d_ws;

  size_t off = 0;
  auto alloc = [&](size_t bytes) {
    size_t o = off;
    off += (bytes + 255) & ~(size_t)255;
    return o;
  };
  unsigned short* FB   = (unsigned short*)(ws + alloc((size_t)MP * 160 * 2)); // bond feats bf16
  float*          BIN  = (float*)         (ws + alloc((size_t)MP * 300 * 4)); // b_input f32
  unsigned short* BMSG = (unsigned short*)(ws + alloc((size_t)MP * 320 * 2)); // b_message bf16
  unsigned short* HPRE = (unsigned short*)(ws + alloc((size_t)MP * 320 * 2)); // pre-GEMM msg bf16
  float*          AMSG = (float*)         (ws + alloc((size_t)AP * 300 * 4)); // a_message f32
  unsigned short* AIN  = (unsigned short*)(ws + alloc((size_t)AP * 448 * 2)); // concat bf16
  unsigned short* WIF  = (unsigned short*)(ws + alloc((size_t)5  * 20 * 512 * 2));
  unsigned short* WHF  = (unsigned short*)(ws + alloc((size_t)10 * 20 * 512 * 2));
  unsigned short* WOF  = (unsigned short*)(ws + alloc((size_t)14 * 20 * 512 * 2));
  int*            CNT  = (int*)           (ws + alloc((size_t)NUM_MOLS * 4));
  // total ~735 MiB of workspace

  dim3 blk(256);
  pack_w_kernel<<<(5 * 20 * 512 + 255) / 256, blk, 0, stream>>>(W_i, WIF, 147, 5);
  pack_w_kernel<<<(10 * 20 * 512 + 255) / 256, blk, 0, stream>>>(W_h, WHF, 300, 10);
  pack_w_kernel<<<(14 * 20 * 512 + 255) / 256, blk, 0, stream>>>(W_o, WOF, 433, 14);
  cvt_bonds_kernel<<<(NBD * 160 + 255) / 256, blk, 0, stream>>>(f_bonds, FB);

  // b_input = f_bonds @ W_i ; b_message = relu(b_input)
  gemm_kernel<0, 5, 5><<<dim3(MP / 128, 4), blk, 0, stream>>>(
      FB, 160, WIF, nullptr, BIN, BMSG, nullptr, NBD);

  for (int d = 0; d < 2; d++) {
    agg_kernel<<<(NAT * 75 + 255) / 256, blk, 0, stream>>>(BMSG, a2b, AMSG);
    bond_pre_kernel<<<(NBD * 80 + 255) / 256, blk, 0, stream>>>(
        AMSG, BMSG, b2a, b2revb, HPRE);
    // b_message = relu(b_input + hpre @ W_h)
    gemm_kernel<1, 10, 5><<<dim3(MP / 128, 4), blk, 0, stream>>>(
        HPRE, 320, WHF, BIN, nullptr, BMSG, nullptr, NBD);
  }

  agg_kernel<<<(NAT * 75 + 255) / 256, blk, 0, stream>>>(BMSG, a2b, AMSG);
  build_ain_kernel<<<(NAT * 448 + 255) / 256, blk, 0, stream>>>(f_atoms, AMSG, AIN);

  zero_kernel<<<(NUM_MOLS * HID + 255) / 256, blk, 0, stream>>>(out, NUM_MOLS * HID);
  zero_kernel<<<(NUM_MOLS + 255) / 256, blk, 0, stream>>>((float*)CNT, NUM_MOLS);

  // atom_hiddens = relu(a_in @ W_o + b_o), segment-summed into out via atomics
  gemm_kernel<2, 14, 4><<<dim3(AP / 128, 5), blk, 0, stream>>>(
      AIN, 448, WOF, b_o, out, nullptr, mol_ids, NAT);

  count_kernel<<<(N_ATOMS + 255) / 256, blk, 0, stream>>>(mol_ids, CNT);
  finalize_kernel<<<(NUM_MOLS * HID + 255) / 256, blk, 0, stream>>>(out, CNT);
}
